// MDNRNN_81664508166708
// MI455X (gfx1250) — matrix operand order repaired and checked by reading, not compiled, not run on hardware
//
#include <hip/hip_runtime.h>
#include <hip/hip_bf16.h>

// ---------------------------------------------------------------------------
// MDN-RNN forward for MI455X (gfx1250): bf16 WMMA GEMMs, wave32,
// double-buffered async LDS staging (global_load_async_to_lds_b128).
// Shapes: B=64, T=50, Z=128, HID=512, G=10, IMG=64 (HW=4096).
// ---------------------------------------------------------------------------

#define BATCH 64
#define TSTEPS 50
#define ZDIM 128
#define HID 512
#define GMIX 10
#define HW 4096

typedef __attribute__((ext_vector_type(16))) __bf16 v16bf;
typedef __attribute__((ext_vector_type(8)))  float  v8f;

union Frag16 { uint4 u[2]; v16bf v; };

// ---------------------------------------------------------------------------
// Generic M=64 GEMM:  C[m][n] = sum_k A[m][k] * W[n][k]  (+ bias, + optional
// accumulate into existing C, + optional sigmoid, + optional bf16 mirror).
// A: bf16, row-major, lda elems.  W: bf16 (N x K) row-major, ldw elems.
// Block = 256 threads = 8 waves tiled 4(M) x 2(N); each wave computes a
// 16x32 output slab (two 16x16 WMMA tiles) -> block covers 64x64 output.
// grid.x = N/64.  K multiple of 32.
// A-tiles (64x32 bf16 = 4KB) are staged through LDS with the CDNA5 async
// copy path (ASYNCcnt) and double buffering so the copy of tile k+1 overlaps
// the WMMA math on tile k.  NOTE: the LDS pointer is passed into the asm as
// a dummy operand so the buffer escapes and LLVM keeps the ds_load reads.
// ---------------------------------------------------------------------------
__launch_bounds__(256)
__global__ void gemm64_wmma_bf16(const __bf16* __restrict__ A, int lda,
                                 const __bf16* __restrict__ W, int ldw,
                                 const float* __restrict__ bias,
                                 float* __restrict__ C, long long ldc,
                                 __bf16* __restrict__ Cb, int ldcb,
                                 int K, int accumulate, int act_sigmoid)
{
    __shared__ __align__(16) __bf16 sA[2][64 * 32];

    const int tid   = threadIdx.x;
    const int lane  = tid & 31;
    const int wv    = tid >> 5;          // wave 0..7
    const int wm    = wv & 3;            // M tile 0..3
    const int wn    = wv >> 2;           // N slab 0..1
    const int lm    = lane & 15;         // 0..15
    const int hi    = lane >> 4;         // 0 or 1
    const int nbase = blockIdx.x * 64 + wn * 32;
    const int ncol0 = nbase + lm;        // output column, subtile 0
    const int ncol1 = nbase + 16 + lm;   // output column, subtile 1

    // per-thread A-staging slice: 16 bytes (8 bf16) of the 64x32 tile
    const int srow = tid >> 2;
    const int scol = (tid & 3) << 3;
    const unsigned ldsOff = (unsigned)(srow * 32 + scol) * 2u; // bytes in buf 0
    const void* ldsEsc = (const void*)&sA[0][0];  // escape LDS buffer into asm

    v8f acc0 = {};
    v8f acc1 = {};

    // prologue: async-copy tile kc=0 into LDS buffer 0
    {
        unsigned long long ga =
            (unsigned long long)(const void*)(A + (long long)srow * lda + scol);
        asm volatile("global_load_async_to_lds_b128 %0, %1, off"
                     :: "v"(ldsOff), "v"(ga), "v"(ldsEsc) : "memory");
    }

    int p = 0;
    for (int kc = 0; kc < K; kc += 32, p ^= 1) {
        // wait for this wave's async copies, then barrier so the whole tile
        // (written by all 8 waves) is visible
        asm volatile("s_wait_asynccnt 0x0" ::: "memory");
        __syncthreads();

        // kick off async copy of the next tile into the other buffer
        if (kc + 32 < K) {
            unsigned long long ga = (unsigned long long)(const void*)(
                A + (long long)srow * lda + kc + 32 + scol);
            unsigned off = ldsOff + (unsigned)((p ^ 1) * 4096);
            asm volatile("global_load_async_to_lds_b128 %0, %1, off"
                         :: "v"(off), "v"(ga), "v"(ldsEsc) : "memory");
        }

        // A fragment (16x32 bf16): lanes 0-15 -> K 0..15; 16-31 -> K 16..31
        Frag16 fa;
        {
            const __bf16* sp = &sA[p][0];
            const int m  = wm * 16 + lm;
            const int b0 = hi * 16;
            fa.u[0] = *reinterpret_cast<const uint4*>(sp + m * 32 + b0);
            fa.u[1] = *reinterpret_cast<const uint4*>(sp + m * 32 + 8 + b0);
        }
        // B fragments (32x16 bf16): lane n, lanes 0-15 K 0..15, 16-31 K 16..31
        Frag16 fb0, fb1;
        {
            const __bf16* wp0 = W + (long long)ncol0 * ldw + kc + hi * 16;
            const __bf16* wp1 = W + (long long)ncol1 * ldw + kc + hi * 16;
            fb0.u[0] = reinterpret_cast<const uint4*>(wp0)[0];
            fb0.u[1] = reinterpret_cast<const uint4*>(wp0)[1];
            fb1.u[0] = reinterpret_cast<const uint4*>(wp1)[0];
            fb1.u[1] = reinterpret_cast<const uint4*>(wp1)[1];
            if (kc + 32 < K) {
                __builtin_prefetch(wp0 + 32, 0, 3);
                __builtin_prefetch(wp1 + 32, 0, 3);
            }
        }
        acc0 = __builtin_amdgcn_wmma_f32_16x16x32_bf16(
                   false, fa.v, false, fb0.v, (short)0, acc0, false, false);
        acc1 = __builtin_amdgcn_wmma_f32_16x16x32_bf16(
                   false, fa.v, false, fb1.v, (short)0, acc1, false, false);
    }

    const float bv0 = bias ? bias[ncol0] : 0.0f;
    const float bv1 = bias ? bias[ncol1] : 0.0f;
#pragma unroll
    for (int r = 0; r < 8; ++r) {
        const int row = wm * 16 + hi * 8 + r;
        const long long i0 = (long long)row * ldc + ncol0;
        const long long i1 = (long long)row * ldc + ncol1;
        float v0 = acc0[r] + bv0;
        float v1 = acc1[r] + bv1;
        if (accumulate) { v0 += C[i0]; v1 += C[i1]; }
        if (act_sigmoid) {
            v0 = 1.0f / (1.0f + __expf(-v0));
            v1 = 1.0f / (1.0f + __expf(-v1));
        }
        C[i0] = v0;
        C[i1] = v1;
        if (Cb) {
            Cb[row * ldcb + ncol0] = (__bf16)v0;
            Cb[row * ldcb + ncol1] = (__bf16)v1;
        }
    }
}

// ---------------------------------------------------------------------------
// f32 -> bf16 weight conversion
// ---------------------------------------------------------------------------
__global__ void f32_to_bf16_kernel(const float* __restrict__ src,
                                   __bf16* __restrict__ dst, int n)
{
    int i = blockIdx.x * blockDim.x + threadIdx.x;
    if (i < n) dst[i] = (__bf16)src[i];
}

// zero the recurrent state (h as bf16 mirror, c as f32). n = BATCH*HID
__global__ void zero_state_kernel(__bf16* __restrict__ hb, float* __restrict__ c)
{
    int i = blockIdx.x * blockDim.x + threadIdx.x;
    hb[i] = (__bf16)0.0f;
    c[i]  = 0.0f;
}

// ---------------------------------------------------------------------------
// MDN head: softmax over G, exp(sigma), write pi/mu/sigma outputs, then draw
// a categorical + gaussian sample -> preds (bf16) for the decoder GEMM.
// One thread per (b, z): 64*128 = 8192 threads.
// ---------------------------------------------------------------------------
__device__ __forceinline__ unsigned pcg_hash(unsigned x)
{
    x = x * 747796405u + 2891336453u;
    unsigned w = ((x >> ((x >> 28) + 4u)) ^ x) * 277803737u;
    return (w >> 22) ^ w;
}

__global__ void mdn_sample_kernel(const float* __restrict__ raw_pi,
                                  const float* __restrict__ raw_mu,
                                  const float* __restrict__ raw_sg,
                                  float* __restrict__ out_pi,
                                  float* __restrict__ out_mu,
                                  float* __restrict__ out_sg,
                                  __bf16* __restrict__ predsb, int t)
{
    const int idx = blockIdx.x * blockDim.x + threadIdx.x;   // 0..8191
    const int b = idx >> 7;
    const int z = idx & (ZDIM - 1);

    float lg[GMIX], mu[GMIX], sg[GMIX];
    float mx = -1e30f;
#pragma unroll
    for (int g = 0; g < GMIX; ++g) {
        const long long o = (long long)b * (GMIX * ZDIM) + g * ZDIM + z;
        lg[g] = raw_pi[o];
        mu[g] = raw_mu[o];
        sg[g] = __expf(raw_sg[o]);
        mx = fmaxf(mx, lg[g]);
    }
    float s = 0.0f;
#pragma unroll
    for (int g = 0; g < GMIX; ++g) { lg[g] = __expf(lg[g] - mx); s += lg[g]; }
    const float inv = 1.0f / s;

    const long long base = ((long long)b * TSTEPS + t) * (GMIX * ZDIM) + z;
#pragma unroll
    for (int g = 0; g < GMIX; ++g) {
        out_pi[base + g * ZDIM] = lg[g] * inv;
        out_mu[base + g * ZDIM] = mu[g];
        out_sg[base + g * ZDIM] = sg[g];
    }

    // deterministic hash RNG (stands in for jax PRNG streams)
    const unsigned h0 = pcg_hash((unsigned)idx * 2654435761u ^ ((unsigned)t * 0x9E3779B9u));
    const unsigned h1 = pcg_hash(h0 ^ 0x85EBCA6Bu);
    const unsigned h2 = pcg_hash(h1 ^ 0x27D4EB2Fu);
    const float u1 = (float)(h0 >> 8) * (1.0f / 16777216.0f);
    const float u2 = fmaxf((float)(h1 >> 8) * (1.0f / 16777216.0f), 1e-7f);
    const float u3 = (float)(h2 >> 8) * (1.0f / 16777216.0f);

    int comp = GMIX - 1;
    float cdf = 0.0f;
#pragma unroll
    for (int g = 0; g < GMIX; ++g) {
        cdf += lg[g] * inv;
        if (u1 < cdf) { comp = g; break; }
    }
    const float eps = sqrtf(-2.0f * __logf(u2)) * __cosf(6.28318530718f * u3);
    predsb[idx] = (__bf16)(mu[comp] + sg[comp] * eps);
}

// ---------------------------------------------------------------------------
// Imputation: input_c = x_t*m_t + img*(1-m_t); writes imputations output and
// bf16 mirror for the encoder GEMM. One thread per (b, pixel): 64*4096.
// ---------------------------------------------------------------------------
__global__ void imputation_kernel(const float* __restrict__ data,
                                  const float* __restrict__ out_pred,
                                  float* __restrict__ out_imp,
                                  __bf16* __restrict__ inputcb, int t)
{
    const int idx = blockIdx.x * blockDim.x + threadIdx.x;  // 0..262143
    const int b = idx >> 12;
    const int i = idx & (HW - 1);
    const long long dbase = (((long long)b * TSTEPS + t) * 2) * HW;
    const float x = data[dbase + i];
    const float m = data[dbase + HW + i];
    const long long pbase = ((long long)b * TSTEPS + t) * HW;
    const float img = out_pred[pbase + i];
    const float v = x * m + img * (1.0f - m);
    out_imp[pbase + i] = v;
    inputcb[idx] = (__bf16)v;
}

// ---------------------------------------------------------------------------
// LSTM pointwise: gates order i,f,g,o. Updates c (f32) and h (bf16 mirror).
// One thread per (b, j): 64*512.
// ---------------------------------------------------------------------------
__global__ void lstm_update_kernel(const float* __restrict__ gates,
                                   float* __restrict__ c,
                                   __bf16* __restrict__ hb)
{
    const int idx = blockIdx.x * blockDim.x + threadIdx.x;  // 0..32767
    const int b = idx >> 9;
    const int j = idx & (HID - 1);
    const float* gr = gates + (long long)b * (4 * HID);
    const float ig = 1.0f / (1.0f + __expf(-gr[j]));
    const float fg = 1.0f / (1.0f + __expf(-gr[HID + j]));
    const float gg = tanhf(gr[2 * HID + j]);
    const float og = 1.0f / (1.0f + __expf(-gr[3 * HID + j]));
    const float c2 = fg * c[idx] + ig * gg;
    c[idx] = c2;
    hb[idx] = (__bf16)(og * tanhf(c2));
}

// ---------------------------------------------------------------------------
// Host launcher
// ---------------------------------------------------------------------------
extern "C" void kernel_launch(void* const* d_in, const int* in_sizes, int n_in,
                              void* d_out, int out_size, void* d_ws, size_t ws_size,
                              hipStream_t stream)
{
    (void)in_sizes; (void)n_in; (void)out_size; (void)ws_size;

    // inputs (setup_inputs order)
    const float* data  = (const float*)d_in[0];
    const float* W_ih  = (const float*)d_in[1];
    const float* W_hh  = (const float*)d_in[2];
    const float* b_ih  = (const float*)d_in[3];
    const float* b_hh  = (const float*)d_in[4];
    const float* fc1_W = (const float*)d_in[5];
    const float* fc1_b = (const float*)d_in[6];
    const float* fc2_W = (const float*)d_in[7];
    const float* fc2_b = (const float*)d_in[8];
    const float* fc3_W = (const float*)d_in[9];
    const float* fc3_b = (const float*)d_in[10];
    const float* dec_W = (const float*)d_in[11];
    const float* dec_b = (const float*)d_in[12];
    const float* enc_W = (const float*)d_in[13];
    const float* enc_b = (const float*)d_in[14];

    // outputs (concatenated flat in return order)
    float* out      = (float*)d_out;
    float* out_pi   = out;                                    // B*T*G*Z
    float* out_mu   = out + (long long)4096000;
    float* out_sg   = out + (long long)8192000;
    float* out_pred = out + (long long)12288000;              // B*T*HW
    float* out_imp  = out + (long long)25395200;              // B*T*HW
    float* out_lat  = out + (long long)38502400;              // B*T*Z

    // workspace layout
    __bf16* wb = (__bf16*)d_ws;
    __bf16* fc1b    = wb;                       // 1280*512
    __bf16* fc2b    = fc1b + 655360;
    __bf16* fc3b    = fc2b + 655360;
    __bf16* decb    = fc3b + 655360;            // 4096*128
    __bf16* encb    = decb + 524288;            // 128*4096
    __bf16* Wihb    = encb + 524288;            // 2048*128
    __bf16* Whhb    = Wihb + 262144;            // 2048*512
    __bf16* hb      = Whhb + 1048576;           // 64*512
    __bf16* predsb  = hb + 32768;               // 64*128
    __bf16* inputcb = predsb + 8192;            // 64*4096
    __bf16* xb      = inputcb + 262144;         // 64*128
    float* fs = (float*)((char*)d_ws + 9273344);
    float* cstate = fs;                          // 64*512
    float* raw1   = cstate + 32768;              // 64*1280
    float* raw2   = raw1 + 81920;
    float* raw3   = raw2 + 81920;
    float* gates  = raw3 + 81920;                // 64*2048

    // --- one-time (per launch, deterministic) weight conversion + state zero
    auto cvt = [&](const float* s, __bf16* d, int n) {
        f32_to_bf16_kernel<<<(n + 255) / 256, 256, 0, stream>>>(s, d, n);
    };
    cvt(fc1_W, fc1b, 655360);
    cvt(fc2_W, fc2b, 655360);
    cvt(fc3_W, fc3b, 655360);
    cvt(dec_W, decb, 524288);
    cvt(enc_W, encb, 524288);
    cvt(W_ih, Wihb, 262144);
    cvt(W_hh, Whhb, 1048576);
    zero_state_kernel<<<(BATCH * HID) / 256, 256, 0, stream>>>(hb, cstate);

    const long long ldc_pred = (long long)TSTEPS * HW;   // 204800
    const long long ldc_lat  = (long long)TSTEPS * ZDIM; // 6400

    for (int t = 0; t < TSTEPS; ++t) {
        // 1) mixture: 3x [64x512]x[512x1280]
        gemm64_wmma_bf16<<<GMIX * ZDIM / 64, 256, 0, stream>>>(
            hb, HID, fc1b, HID, fc1_b, raw1, GMIX * ZDIM, nullptr, 0, HID, 0, 0);
        gemm64_wmma_bf16<<<GMIX * ZDIM / 64, 256, 0, stream>>>(
            hb, HID, fc2b, HID, fc2_b, raw2, GMIX * ZDIM, nullptr, 0, HID, 0, 0);
        gemm64_wmma_bf16<<<GMIX * ZDIM / 64, 256, 0, stream>>>(
            hb, HID, fc3b, HID, fc3_b, raw3, GMIX * ZDIM, nullptr, 0, HID, 0, 0);

        // 2) softmax/exp + sampling -> pi/mu/sigma outputs, preds (bf16)
        mdn_sample_kernel<<<(BATCH * ZDIM) / 256, 256, 0, stream>>>(
            raw1, raw2, raw3, out_pi, out_mu, out_sg, predsb, t);

        // 3) decoder [64x128]x[128x4096] + sigmoid -> predicts output
        gemm64_wmma_bf16<<<HW / 64, 256, 0, stream>>>(
            predsb, ZDIM, decb, ZDIM, dec_b, out_pred + (long long)t * HW,
            ldc_pred, nullptr, 0, ZDIM, 0, 1);

        // 4) imputation -> imputations output + bf16 mirror
        imputation_kernel<<<(BATCH * HW) / 256, 256, 0, stream>>>(
            data, out_pred, out_imp, inputcb, t);

        // 5) encoder [64x4096]x[4096x128] -> latents output + xb mirror
        gemm64_wmma_bf16<<<ZDIM / 64, 256, 0, stream>>>(
            inputcb, HW, encb, HW, enc_b, out_lat + (long long)t * ZDIM,
            ldc_lat, xb, ZDIM, HW, 0, 0);

        // 6) gates = x@W_ih^T + b_ih  (then += h@W_hh^T + b_hh)
        gemm64_wmma_bf16<<<(4 * HID) / 64, 256, 0, stream>>>(
            xb, ZDIM, Wihb, ZDIM, b_ih, gates, 4 * HID, nullptr, 0, ZDIM, 0, 0);
        gemm64_wmma_bf16<<<(4 * HID) / 64, 256, 0, stream>>>(
            hb, HID, Whhb, HID, b_hh, gates, 4 * HID, nullptr, 0, HID, 1, 0);

        // 7) LSTM pointwise update of (h, c)
        lstm_update_kernel<<<(BATCH * HID) / 256, 256, 0, stream>>>(
            gates, cstate, hb);
    }
}
